// MambaFusionBlock_55886114456110
// MI455X (gfx1250) — matrix operand order repaired and checked by reading, not compiled, not run on hardware
//
#include <hip/hip_runtime.h>
#include <hip/hip_bf16.h>
#include <math.h>
#include <stdint.h>

// ---------------- problem constants ----------------
#define BATCH   4
#define CCH     96
#define HH      96
#define WW      96
#define LSEQ    (HH * WW)          // 9216
#define DIN     192                // d_inner
#define DST     16                 // d_state
#define DTR     6                  // dt_rank
#define MROWS   (BATCH * LSEQ)     // 36864
#define NXZ     (2 * DIN)          // 384
#define NXP     48                 // x_proj N padded (38 -> 48)
#define NCHUNK  16
#define CLEN    (LSEQ / NCHUNK)    // 576
#define LN_EPS  1e-5f
#define KMAX    256                // max K staged in LDS (we use 192)

typedef __bf16 bf16_t;
typedef __attribute__((ext_vector_type(8)))  __bf16 v8bf;
typedef __attribute__((ext_vector_type(16))) __bf16 v16bf;
typedef __attribute__((ext_vector_type(8)))  float  v8f;
typedef unsigned int u32;
typedef __attribute__((ext_vector_type(4)))  u32 v4u;
typedef __attribute__((ext_vector_type(8)))  int v8i;
typedef __attribute__((ext_vector_type(4)))  int v4i;

#if defined(__has_builtin)
#  if __has_builtin(__builtin_amdgcn_tensor_load_to_lds) && \
      __has_builtin(__builtin_amdgcn_s_wait_tensorcnt)
#    define HAVE_TDM 1
#  endif
#endif

__device__ __forceinline__ float silu_f(float x) {
    return x / (1.0f + __expf(-x));
}
__device__ __forceinline__ float softplus_f(float x) {
    return (x > 20.0f) ? x : log1pf(__expf(x));
}

// ---------------------------------------------------------------------------
// prep: fold proj into in_proj (W2 = in_proj_w @ proj_w, bf16), fold bias,
// pad/convert x_proj_w & out_proj_w to bf16, compute A = -exp(A_log).
// ---------------------------------------------------------------------------
__global__ void mf_prep_kernel(const float* __restrict__ proj_w,
                               const float* __restrict__ proj_b,
                               const float* __restrict__ in_proj_w,
                               const float* __restrict__ x_proj_w,
                               const float* __restrict__ out_proj_w,
                               const float* __restrict__ A_log,
                               bf16_t* __restrict__ W2, float* __restrict__ b2,
                               bf16_t* __restrict__ xpw, bf16_t* __restrict__ opw,
                               float* __restrict__ Aneg) {
    int id = blockIdx.x * blockDim.x + threadIdx.x;
    const int nW2 = NXZ * DIN;
    const int nB2 = NXZ;
    const int nXP = NXP * DIN;
    const int nOP = CCH * DIN;
    const int nA  = DIN * DST;
    if (id >= nW2 + nB2 + nXP + nOP + nA) return;
    if (id < nW2) {
        int o = id / DIN, i = id % DIN;
        float acc = 0.f;
        #pragma unroll 8
        for (int k = 0; k < CCH; ++k)
            acc += in_proj_w[o * CCH + k] * proj_w[k * DIN + i];
        W2[o * DIN + i] = (bf16_t)acc;
    } else if ((id -= nW2) < nB2) {
        float acc = 0.f;
        for (int k = 0; k < CCH; ++k)
            acc += in_proj_w[id * CCH + k] * proj_b[k];
        b2[id] = acc;
    } else if ((id -= nB2) < nXP) {
        int n = id / DIN, k = id % DIN;
        xpw[id] = (n < (DTR + 2 * DST)) ? (bf16_t)x_proj_w[n * DIN + k]
                                        : (bf16_t)0.0f;
    } else if ((id -= nXP) < nOP) {
        opw[id] = (bf16_t)out_proj_w[id];
    } else {
        id -= nOP;
        Aneg[id] = -__expf(A_log[id]);
    }
}

// ---------------------------------------------------------------------------
// pack seq: concat(spatial,freq) [B,2C,L] -> row-major bf16 [M][2C]
// ---------------------------------------------------------------------------
__global__ void mf_pack_seq_kernel(const float* __restrict__ spat,
                                   const float* __restrict__ freq,
                                   bf16_t* __restrict__ xT) {
    int id = blockIdx.x * blockDim.x + threadIdx.x;
    if (id >= BATCH * (2 * CCH) * LSEQ) return;
    int l = id % LSEQ;
    int c = (id / LSEQ) % (2 * CCH);
    int b = id / (LSEQ * 2 * CCH);
    float v = (c < CCH) ? spat[(size_t)(b * CCH + c) * LSEQ + l]
                        : freq[(size_t)(b * CCH + (c - CCH)) * LSEQ + l];
    xT[((size_t)b * LSEQ + l) * (2 * CCH) + c] = (bf16_t)v;
}

// ---------------------------------------------------------------------------
// WMMA bf16 GEMM:  C[M][N] = A[M][K] * W[N][K]^T + bias
// Block = 4 waves, all sharing one 16-row weight slab (same tn). The slab
// (16 x K bf16, <= 8KB) is staged into LDS by the Tensor Data Mover once per
// block; activations stream from global. One 16x16 tile per wave, f32 accum.
// Requires: tilesM % 4 == 0 and total tiles % 4 == 0 (true for all call sites).
// ---------------------------------------------------------------------------
__global__ void mf_gemm_bf16_kernel(const bf16_t* __restrict__ A,
                                    const bf16_t* __restrict__ Wt,
                                    const float* __restrict__ bias,
                                    float* __restrict__ Cout,
                                    int M, int N, int K) {
    __shared__ __align__(64) bf16_t ldsW[16 * KMAX];
    const int lane = threadIdx.x & 31;
    const int wid  = threadIdx.x >> 5;
    const int tilesM = M >> 4;
    const int blockTile0 = blockIdx.x * 4;
    const int tn = blockTile0 / tilesM;          // shared by all 4 waves
    const int tm = (blockTile0 % tilesM) + wid;
    const int hi   = (lane < 16) ? 0 : 1;        // which K-half this lane owns
    const int lrow = lane & 15;

#ifdef HAVE_TDM
    // ---- TDM: stage 16 x K weight slab into LDS (descriptor per ISA ch.8) ----
    if (wid == 0) {
        uint64_t ga = (uint64_t)(uintptr_t)(Wt + (size_t)tn * 16 * K);
        uint32_t lds_off = (uint32_t)(uintptr_t)(&ldsW[0]);   // addr[31:0] = LDS byte offset
        v4u g0 = { 1u,                                        // count=1 (valid, user mode)
                   lds_off,                                   // lds_addr
                   (u32)ga,                                   // global_addr[31:0]
                   (u32)((ga >> 32) & 0x1FFFFFFu) | (2u << 30) }; // addr[56:32] | type=2
        u32 td0 = (u32)K;                                     // tensor_dim0 (elements)
        u32 td1 = 16u;                                        // tensor_dim1 (rows)
        v8i g1 = { (int)(1u << 16),                                   // data_size=2B, mask=0
                   (int)((td0 & 0xFFFFu) << 16),                      // tensor_dim0[15:0]
                   (int)(((td0 >> 16) & 0xFFFFu) | ((td1 & 0xFFFFu) << 16)),
                   (int)(((td1 >> 16) & 0xFFFFu) | (((u32)K & 0xFFFFu) << 16)), // tile_dim0=K
                   (int)16,                                           // tile_dim1=16, tile_dim2=0
                   (int)(u32)K,                                       // dim0_stride[31:0]
                   0, 0 };                                            // stride hi / dim1_stride
        v4i gz = { 0, 0, 0, 0 };
#if defined(__clang_major__) && (__clang_major__ >= 23)
        v8i gz8 = { 0, 0, 0, 0, 0, 0, 0, 0 };
        __builtin_amdgcn_tensor_load_to_lds(g0, g1, gz, gz, gz8, 0);
#else
        __builtin_amdgcn_tensor_load_to_lds(g0, g1, gz, gz, 0);
#endif
        __builtin_amdgcn_s_wait_tensorcnt(0);
    }
    __syncthreads();
    const bf16_t* brow = &ldsW[lrow * K + hi * 16];
#else
    const bf16_t* brow = Wt + (size_t)(tn * 16 + lrow) * K + hi * 16;
#endif

    const bf16_t* arow = A + (size_t)(tm * 16 + lrow) * K + hi * 16;

    v8f acc = {};
    for (int kk = 0; kk < K; kk += 32) {
        if (kk + 32 < K)   // speculative prefetch of next activation K-slab
            __builtin_prefetch(arow + kk + 32, 0, 1);
        v8bf alo = *(const v8bf*)(arow + kk);        // K = base+{0..7}/{16..23}
        v8bf ahi = *(const v8bf*)(arow + kk + 8);    // K = base+{8..15}/{24..31}
        v16bf afrag;
        v16bf bfrag = *(const v16bf*)(brow + kk);    // K = base+{0..15}/{16..31}
        #pragma unroll
        for (int i = 0; i < 8; ++i) { afrag[i] = alo[i]; afrag[i + 8] = ahi[i]; }
        acc = __builtin_amdgcn_wmma_f32_16x16x32_bf16(
                  false, afrag, false, bfrag, (short)0, acc, false, false);
    }
    const int col = tn * 16 + lrow;
    const int rowbase = tm * 16 + hi * 8;
    const float bv = bias ? bias[col] : 0.0f;
    #pragma unroll
    for (int r = 0; r < 8; ++r)
        Cout[(size_t)(rowbase + r) * N + col] = acc[r] + bv;
}

// ---------------------------------------------------------------------------
// causal depthwise conv (k=4) + bias + SiLU; write f32 and bf16 copies
// ---------------------------------------------------------------------------
__global__ void mf_conv_silu_kernel(const float* __restrict__ xz,
                                    const float* __restrict__ conv_w,
                                    const float* __restrict__ conv_b,
                                    float* __restrict__ xc,
                                    bf16_t* __restrict__ xcb) {
    int id = blockIdx.x * blockDim.x + threadIdx.x;
    if (id >= MROWS * DIN) return;
    int d = id % DIN;
    int m = id / DIN;
    int b = m / LSEQ, l = m % LSEQ;
    float acc = conv_b[d];
    #pragma unroll
    for (int j = 0; j < 4; ++j) {
        int li = l - 3 + j;
        if (li >= 0)
            acc += conv_w[d * 4 + j] * xz[((size_t)(b * LSEQ + li)) * NXZ + d];
    }
    float v = silu_f(acc);
    xc[(size_t)m * DIN + d]  = v;
    xcb[(size_t)m * DIN + d] = (bf16_t)v;
}

// ---------------------------------------------------------------------------
// delta = softplus(dt @ dt_proj_w^T + dt_proj_b)   (K=6 -> plain VALU)
// ---------------------------------------------------------------------------
__global__ void mf_delta_kernel(const float* __restrict__ xdbl,
                                const float* __restrict__ dt_proj_w,
                                const float* __restrict__ dt_proj_b,
                                float* __restrict__ delta) {
    int id = blockIdx.x * blockDim.x + threadIdx.x;
    if (id >= MROWS * DIN) return;
    int d = id % DIN;
    int m = id / DIN;
    float acc = dt_proj_b[d];
    #pragma unroll
    for (int r = 0; r < DTR; ++r)
        acc += xdbl[(size_t)m * NXP + r] * dt_proj_w[d * DTR + r];
    delta[(size_t)m * DIN + d] = softplus_f(acc);
}

// ---------------------------------------------------------------------------
// Scan pass 1: per (b, chunk, d) channel (16 lanes = 16 states):
//   P = prod(exp(delta*A)),  F = local final state with h0 = 0
// ---------------------------------------------------------------------------
__global__ void mf_scan1_kernel(const float* __restrict__ delta,
                                const float* __restrict__ xc,
                                const float* __restrict__ xdbl,
                                const float* __restrict__ Aneg,
                                float* __restrict__ Pbuf,
                                float* __restrict__ Fbuf) {
    int gwave = (blockIdx.x * blockDim.x + threadIdx.x) >> 5;
    int lane = threadIdx.x & 31;
    int s = lane & 15, sub = lane >> 4;
    int b = gwave / (NCHUNK * (DIN / 2));
    int rem = gwave % (NCHUNK * (DIN / 2));
    int chunk = rem / (DIN / 2);
    int d = (rem % (DIN / 2)) * 2 + sub;
    if (b >= BATCH) return;

    const float a_coef = Aneg[d * DST + s];
    float h = 0.0f, P = 1.0f;
    int m0 = b * LSEQ + chunk * CLEN;
    for (int t = 0; t < CLEN; ++t) {
        size_t m = (size_t)(m0 + t);
        float dl = delta[m * DIN + d];
        float xv = xc[m * DIN + d];
        float bm = xdbl[m * NXP + DTR + s];
        float a = __expf(dl * a_coef);
        h = a * h + dl * xv * bm;
        P *= a;
    }
    size_t idx = (((size_t)b * DIN + d) * DST + s) * NCHUNK + chunk;
    Pbuf[idx] = P;
    Fbuf[idx] = h;
}

// ---------------------------------------------------------------------------
// Combine: sequential scan over the 16 chunks per (b,d,s) -> chunk h0's
// ---------------------------------------------------------------------------
__global__ void mf_combine_kernel(const float* __restrict__ Pbuf,
                                  const float* __restrict__ Fbuf,
                                  float* __restrict__ H0buf) {
    int id = blockIdx.x * blockDim.x + threadIdx.x;
    if (id >= BATCH * DIN * DST) return;
    size_t base = (size_t)id * NCHUNK;
    float h = 0.0f;
    #pragma unroll
    for (int c = 0; c < NCHUNK; ++c) {
        H0buf[base + c] = h;
        h = Pbuf[base + c] * h + Fbuf[base + c];
    }
}

// ---------------------------------------------------------------------------
// Scan pass 2: replay chunk with correct h0, emit gated y (bf16 for out_proj)
//   y = (sum_s h*C + xc*D) * silu(z)
// ---------------------------------------------------------------------------
__global__ void mf_scan2_kernel(const float* __restrict__ delta,
                                const float* __restrict__ xc,
                                const float* __restrict__ xdbl,
                                const float* __restrict__ Aneg,
                                const float* __restrict__ H0buf,
                                const float* __restrict__ xz,
                                const float* __restrict__ D_param,
                                bf16_t* __restrict__ yg) {
    int gwave = (blockIdx.x * blockDim.x + threadIdx.x) >> 5;
    int lane = threadIdx.x & 31;
    int s = lane & 15, sub = lane >> 4;
    int b = gwave / (NCHUNK * (DIN / 2));
    int rem = gwave % (NCHUNK * (DIN / 2));
    int chunk = rem / (DIN / 2);
    int d = (rem % (DIN / 2)) * 2 + sub;
    if (b >= BATCH) return;

    const float a_coef = Aneg[d * DST + s];
    const float dpar = D_param[d];
    float h = H0buf[(((size_t)b * DIN + d) * DST + s) * NCHUNK + chunk];
    int m0 = b * LSEQ + chunk * CLEN;
    for (int t = 0; t < CLEN; ++t) {
        size_t m = (size_t)(m0 + t);
        float dl = delta[m * DIN + d];
        float xv = xc[m * DIN + d];
        float bm = xdbl[m * NXP + DTR + s];
        float cm = xdbl[m * NXP + DTR + DST + s];
        float a = __expf(dl * a_coef);
        h = a * h + dl * xv * bm;
        float yp = h * cm;
        #pragma unroll
        for (int off = 1; off < 16; off <<= 1)   // XOR<16 stays inside state group
            yp += __shfl_xor(yp, off, 32);
        if (s == 0) {
            float z = xz[m * NXZ + DIN + d];
            float yfull = (yp + xv * dpar) * silu_f(z);
            yg[m * DIN + d] = (bf16_t)yfull;
        }
    }
}

// ---------------------------------------------------------------------------
// LayerNorm over channels + residual + [B,L,C] -> [B,C,H,W]
// ---------------------------------------------------------------------------
__global__ void mf_ln_out_kernel(const float* __restrict__ mixed,
                                 const float* __restrict__ spat,
                                 const float* __restrict__ ln_g,
                                 const float* __restrict__ ln_b,
                                 float* __restrict__ out) {
    int m = blockIdx.x * blockDim.x + threadIdx.x;
    if (m >= MROWS) return;
    const float* row = mixed + (size_t)m * CCH;
    float sum = 0.f, sumsq = 0.f;
    #pragma unroll 8
    for (int c = 0; c < CCH; ++c) { float v = row[c]; sum += v; sumsq += v * v; }
    float mean = sum * (1.0f / CCH);
    float var = sumsq * (1.0f / CCH) - mean * mean;
    float inv = rsqrtf(var + LN_EPS);
    int b = m / LSEQ, l = m % LSEQ;
    for (int c = 0; c < CCH; ++c) {
        size_t oidx = (size_t)(b * CCH + c) * LSEQ + l;
        float nv = (row[c] - mean) * inv * ln_g[c] + ln_b[c];
        out[oidx] = nv + spat[oidx];
    }
}

// ---------------------------------------------------------------------------
extern "C" void kernel_launch(void* const* d_in, const int* in_sizes, int n_in,
                              void* d_out, int out_size, void* d_ws, size_t ws_size,
                              hipStream_t stream) {
    const float* spatial   = (const float*)d_in[0];
    const float* freq      = (const float*)d_in[1];
    const float* proj_w    = (const float*)d_in[2];
    const float* proj_b    = (const float*)d_in[3];
    const float* in_proj_w = (const float*)d_in[4];
    const float* conv_w    = (const float*)d_in[5];
    const float* conv_b    = (const float*)d_in[6];
    const float* x_proj_w  = (const float*)d_in[7];
    const float* dt_proj_w = (const float*)d_in[8];
    const float* dt_proj_b = (const float*)d_in[9];
    const float* A_log     = (const float*)d_in[10];
    const float* D_param   = (const float*)d_in[11];
    const float* out_proj_w= (const float*)d_in[12];
    const float* ln_g      = (const float*)d_in[13];
    const float* ln_b      = (const float*)d_in[14];
    float* out = (float*)d_out;

    char* ws = (char*)d_ws;
    size_t cur = 0;
    auto walloc = [&](size_t bytes) -> char* {
        char* p = ws + cur;
        cur = (cur + bytes + 255) & ~(size_t)255;
        return p;
    };
    bf16_t* xT    = (bf16_t*)walloc((size_t)MROWS * 2 * CCH * sizeof(bf16_t));
    bf16_t* W2    = (bf16_t*)walloc((size_t)NXZ * DIN * sizeof(bf16_t));
    float*  b2    = (float*) walloc((size_t)NXZ * sizeof(float));
    bf16_t* xpw   = (bf16_t*)walloc((size_t)NXP * DIN * sizeof(bf16_t));
    bf16_t* opw   = (bf16_t*)walloc((size_t)CCH * DIN * sizeof(bf16_t));
    float*  Aneg  = (float*) walloc((size_t)DIN * DST * sizeof(float));
    float*  xz    = (float*) walloc((size_t)MROWS * NXZ * sizeof(float));
    float*  xc    = (float*) walloc((size_t)MROWS * DIN * sizeof(float));
    bf16_t* xcb   = (bf16_t*)walloc((size_t)MROWS * DIN * sizeof(bf16_t));
    float*  xdbl  = (float*) walloc((size_t)MROWS * NXP * sizeof(float));
    float*  delta = (float*) walloc((size_t)MROWS * DIN * sizeof(float));
    float*  Pbuf  = (float*) walloc((size_t)BATCH * DIN * DST * NCHUNK * sizeof(float));
    float*  Fbuf  = (float*) walloc((size_t)BATCH * DIN * DST * NCHUNK * sizeof(float));
    float*  H0buf = (float*) walloc((size_t)BATCH * DIN * DST * NCHUNK * sizeof(float));
    bf16_t* yg    = (bf16_t*)walloc((size_t)MROWS * DIN * sizeof(bf16_t));
    float*  mixed = (float*) walloc((size_t)MROWS * CCH * sizeof(float));
    (void)ws_size; (void)in_sizes; (void)n_in; (void)out_size;

    const int TB = 256;
    // 1) weight folding / conversion
    {
        int total = NXZ * DIN + NXZ + NXP * DIN + CCH * DIN + DIN * DST;
        mf_prep_kernel<<<(total + TB - 1) / TB, TB, 0, stream>>>(
            proj_w, proj_b, in_proj_w, x_proj_w, out_proj_w, A_log,
            W2, b2, xpw, opw, Aneg);
    }
    // 2) pack concat(spatial,freq) into row-major bf16 [M][192]
    {
        int total = BATCH * 2 * CCH * LSEQ;
        mf_pack_seq_kernel<<<(total + TB - 1) / TB, TB, 0, stream>>>(spatial, freq, xT);
    }
    // 3) fused proj+in_proj GEMM: xz[M][384]
    {
        int tiles = (MROWS / 16) * (NXZ / 16);       // divisible by 4
        mf_gemm_bf16_kernel<<<tiles / 4, 128, 0, stream>>>(
            xT, W2, b2, xz, MROWS, NXZ, 2 * CCH);
    }
    // 4) causal depthwise conv + SiLU
    {
        int total = MROWS * DIN;
        mf_conv_silu_kernel<<<(total + TB - 1) / TB, TB, 0, stream>>>(
            xz, conv_w, conv_b, xc, xcb);
    }
    // 5) x_proj GEMM: xdbl[M][48] (cols 0..37 valid)
    {
        int tiles = (MROWS / 16) * (NXP / 16);
        mf_gemm_bf16_kernel<<<tiles / 4, 128, 0, stream>>>(
            xcb, xpw, nullptr, xdbl, MROWS, NXP, DIN);
    }
    // 6) delta
    {
        int total = MROWS * DIN;
        mf_delta_kernel<<<(total + TB - 1) / TB, TB, 0, stream>>>(
            xdbl, dt_proj_w, dt_proj_b, delta);
    }
    // 7) chunked selective scan, 3 phases
    {
        int waves = BATCH * NCHUNK * (DIN / 2);             // 6144
        int blocks = (waves * 32 + TB - 1) / TB;            // 8 waves/block
        mf_scan1_kernel<<<blocks, TB, 0, stream>>>(delta, xc, xdbl, Aneg, Pbuf, Fbuf);
        int nch = BATCH * DIN * DST;                        // 12288
        mf_combine_kernel<<<(nch + TB - 1) / TB, TB, 0, stream>>>(Pbuf, Fbuf, H0buf);
        mf_scan2_kernel<<<blocks, TB, 0, stream>>>(delta, xc, xdbl, Aneg, H0buf,
                                                   xz, D_param, yg);
    }
    // 8) out_proj GEMM: mixed[M][96]
    {
        int tiles = (MROWS / 16) * (CCH / 16);
        mf_gemm_bf16_kernel<<<tiles / 4, 128, 0, stream>>>(
            yg, opw, nullptr, mixed, MROWS, CCH, DIN);
    }
    // 9) LayerNorm + residual + layout restore
    {
        mf_ln_out_kernel<<<(MROWS + TB - 1) / TB, TB, 0, stream>>>(
            mixed, spatial, ln_g, ln_b, out);
    }
}
